// UMA_62105227100396
// MI455X (gfx1250) — compile-verified, matrix operand order checked by reading, not run on hardware
//
#include <hip/hip_runtime.h>
#include <hip/hip_bf16.h>

// ---------------- problem constants ----------------
#define Bb    8
#define Ll    2048
#define Dd    256
#define MAXC  (Ll / 2 + 1)   // 1025

typedef __bf16 bf16_t;
typedef __attribute__((ext_vector_type(16))) __bf16 v16bf;
typedef __attribute__((ext_vector_type(8)))  float  v8f;

// WMMA 16x16x32 bf16 fragment loader (ISA 7.12.2 16-bit layout):
//   lanes 0-15  : K = colbase+{0..7} and colbase+{16..23}
//   lanes 16-31 : K = colbase+{8..15} and colbase+{24..31}
// Row index (A: M, B: N) = rowbase + (lane & 15). Row-major source, ld = Dd.
static __device__ __forceinline__ v16bf load_frag_bf16(const bf16_t* base,
                                                       int row, int ld, int colbase,
                                                       int lane) {
    const bf16_t* p = base + (size_t)row * ld + colbase + ((lane & 16) ? 8 : 0);
    union { uint4 u[2]; v16bf v; } cv;
    cv.u[0] = *(const uint4*)(p);        // 8 bf16, 16 bytes
    cv.u[1] = *(const uint4*)(p + 16);   // next K-half
    return cv.v;
}

// C/D tile layout: VGPR v -> row = v + ((lane&16)>>1), col = lane & 15
static __device__ __forceinline__ int c_row(int v, int lane) { return v + ((lane & 16) >> 1); }

// =====================================================================
// Kernel 0: one-time f32 -> bf16 conversion (native v_cvt, packed stores)
// =====================================================================
__global__ void cvt_to_bf16(const float* __restrict__ src,
                            bf16_t* __restrict__ dst, int n4) {
    int i = (blockIdx.x * blockDim.x + threadIdx.x);
    if (i >= n4) return;
    float4 f = ((const float4*)src)[i];
    union { bf16_t h[4]; unsigned long long u; } cv;
    cv.h[0] = (bf16_t)f.x; cv.h[1] = (bf16_t)f.y;
    cv.h[2] = (bf16_t)f.z; cv.h[3] = (bf16_t)f.w;
    ((unsigned long long*)dst)[i] = cv.u;
}

// =====================================================================
// Kernel 1: Q = x @ wq^T + bq ; K = x @ wk^T + bk   (all bf16 operands)
// grid = (B*L)/16 WGs, 256 threads (8 waves). Wave w does 4 of 32 jobs;
// the 8 A-fragments (x-tile) live in registers, shared across its jobs.
// =====================================================================
__global__ void uma_proj_qk(const bf16_t* __restrict__ Xb,
                            const bf16_t* __restrict__ Wqb, const float* __restrict__ bq,
                            const bf16_t* __restrict__ Wkb, const float* __restrict__ bk,
                            bf16_t* __restrict__ Qb,
                            bf16_t* __restrict__ Kb) {
    const int tid  = threadIdx.x;
    const int lane = tid & 31;
    const int wave = tid >> 5;
    const int rbase = blockIdx.x * 16;                // row in [0, B*L)

    // preload A-fragments of the x-tile (reused by all 4 jobs of this wave)
    v16bf a[Dd / 32];
#pragma unroll
    for (int kk = 0; kk < Dd / 32; ++kk)
        a[kk] = load_frag_bf16(Xb, rbase + (lane & 15), Dd, kk * 32, lane);

#pragma unroll
    for (int j = 0; j < 4; ++j) {
        const int job    = wave * 4 + j;              // 0..31
        const int is_k   = job >> 4;                  // 0 -> Q, 1 -> K
        const int nbase  = (job & 15) * 16;
        const bf16_t* w  = is_k ? Wkb : Wqb;
        const float* bia = is_k ? bk : bq;

        v8f acc = {};
#pragma unroll
        for (int kk = 0; kk < Dd / 32; ++kk) {
            // B[K=d][N=n] = w[n*Dd + d]  (row-major w, contiguous along d)
            v16bf bfr = load_frag_bf16(w, nbase + (lane & 15), Dd, kk * 32, lane);
            acc = __builtin_amdgcn_wmma_f32_16x16x32_bf16(false, a[kk], false, bfr,
                                                          (short)0, acc, false, false);
        }
        const float bias = bia[nbase + (lane & 15)];
        bf16_t* dst = is_k ? Kb : Qb;
#pragma unroll
        for (int v = 0; v < 8; ++v) {
            int row = rbase + c_row(v, lane);
            int col = nbase + (lane & 15);
            dst[(size_t)row * Dd + col] = (bf16_t)(acc[v] + bias);
        }
    }
}

// =====================================================================
// Kernel 2: softmax denominators  S[b,q] = sum_{k<olen} exp(score(q,k))
// grid = B*(L/16), 256 threads; wave w covers k-tiles [w*16, w*16+16)
// =====================================================================
__global__ void uma_rowsum(const bf16_t* __restrict__ Qb,
                           const bf16_t* __restrict__ Kb,
                           const int* __restrict__ olens,
                           const float* __restrict__ gw,
                           float* __restrict__ S) {
    __shared__ float rs[16];
    const int tid  = threadIdx.x;
    const int lane = tid & 31;
    const int wave = tid >> 5;
    const int b     = blockIdx.x >> 7;
    const int qbase = (blockIdx.x & 127) * 16;
    const int olen  = olens[b];
    const float g   = gw[0];
    const float scale = 0.0625f;                      // 1/sqrt(256)

    const bf16_t* Qrow = Qb + (size_t)b * Ll * Dd;
    const bf16_t* Krow = Kb + (size_t)b * Ll * Dd;

    // preload the 8 A-fragments of this q-tile (reused for all 128 k-tiles)
    v16bf a[Dd / 32];
#pragma unroll
    for (int kk = 0; kk < Dd / 32; ++kk)
        a[kk] = load_frag_bf16(Qrow, qbase + (lane & 15), Dd, kk * 32, lane);

    float rowpart[8];
#pragma unroll
    for (int v = 0; v < 8; ++v) rowpart[v] = 0.f;

    for (int t = 0; t < 16; ++t) {
        const int kbase = (wave * 16 + t) * 16;
        __builtin_prefetch(Krow + (size_t)(kbase + 16) * Dd, 0, 1);
        v8f acc = {};
#pragma unroll
        for (int kk = 0; kk < Dd / 32; ++kk) {
            v16bf bfr = load_frag_bf16(Krow, kbase + (lane & 15), Dd, kk * 32, lane);
            acc = __builtin_amdgcn_wmma_f32_16x16x32_bf16(false, a[kk], false, bfr,
                                                          (short)0, acc, false, false);
        }
        const int k = kbase + (lane & 15);
#pragma unroll
        for (int v = 0; v < 8; ++v) {
            int q = qbase + c_row(v, lane);
            float d = (float)(q - k);
            float s = acc[v] * scale - fabsf(g * d * d);
            if (k < olen) rowpart[v] += __expf(s);
        }
    }

    if (tid < 16) rs[tid] = 0.f;
    __syncthreads();
#pragma unroll
    for (int v = 0; v < 8; ++v) atomicAdd(&rs[c_row(v, lane)], rowpart[v]);  // ds_add_f32
    __syncthreads();
    if (tid < 16) S[(size_t)b * Ll + qbase + tid] = rs[tid];
}

// =====================================================================
// Kernel 3: u[b,k] = sum_q exp(score)/S[q]  (each WG owns 16 exclusive k cols)
// grid = B*(L/16), 256 threads; wave w covers q-tiles [w*16, w*16+16)
// =====================================================================
__global__ void uma_colsum(const bf16_t* __restrict__ Qb,
                           const bf16_t* __restrict__ Kb,
                           const int* __restrict__ olens,
                           const float* __restrict__ gw,
                           const float* __restrict__ S,
                           float* __restrict__ U) {
    __shared__ float cs[16];
    const int tid  = threadIdx.x;
    const int lane = tid & 31;
    const int wave = tid >> 5;
    const int b     = blockIdx.x >> 7;
    const int kbase = (blockIdx.x & 127) * 16;
    const int olen  = olens[b];
    const float g   = gw[0];
    const float scale = 0.0625f;

    const bf16_t* Qrow = Qb + (size_t)b * Ll * Dd;
    const bf16_t* Krow = Kb + (size_t)b * Ll * Dd;
    const float* Srow = S + (size_t)b * Ll;

    // preload the 8 B-fragments of this k-tile (reused for all 128 q-tiles)
    v16bf bfr[Dd / 32];
#pragma unroll
    for (int kk = 0; kk < Dd / 32; ++kk)
        bfr[kk] = load_frag_bf16(Krow, kbase + (lane & 15), Dd, kk * 32, lane);

    const int k = kbase + (lane & 15);
    const bool kvalid = (k < olen);
    float colpart = 0.f;

    for (int t = 0; t < 16; ++t) {
        const int qbase = (wave * 16 + t) * 16;
        __builtin_prefetch(Qrow + (size_t)(qbase + 16) * Dd, 0, 1);
        v8f acc = {};
#pragma unroll
        for (int kk = 0; kk < Dd / 32; ++kk) {
            v16bf a = load_frag_bf16(Qrow, qbase + (lane & 15), Dd, kk * 32, lane);
            acc = __builtin_amdgcn_wmma_f32_16x16x32_bf16(false, a, false, bfr[kk],
                                                          (short)0, acc, false, false);
        }
        if (kvalid) {
#pragma unroll
            for (int v = 0; v < 8; ++v) {
                int q = qbase + c_row(v, lane);
                float d = (float)(q - k);
                float s = acc[v] * scale - fabsf(g * d * d);
                colpart += __expf(s) / fmaxf(Srow[q], 1e-30f);
            }
        }
    }

    if (tid < 16) cs[tid] = 0.f;
    __syncthreads();
    atomicAdd(&cs[lane & 15], colpart);               // ds_add_f32
    __syncthreads();
    if (tid < 16) U[(size_t)b * Ll + kbase + tid] = cs[tid];
}

// =====================================================================
// Kernel 4a: valley scan -> segment starts/ends + counts (serial, tiny)
// =====================================================================
__global__ void uma_valleys(const float* __restrict__ U,
                            int* __restrict__ starts, int* __restrict__ ends,
                            float* __restrict__ countsOut) {
    const int b = blockIdx.x;
    if (threadIdx.x != 0) return;
    const float* u = U + (size_t)b * Ll;
    int cs = 0, ce = 0, total = 0;
    for (int l = 0; l < Ll; ++l) {
        float before = (l > 0)      ? u[l - 1] : 0.f;
        float after  = (l < Ll - 1) ? u[l + 1] : 0.f;
        bool valley = (u[l] < before) && (u[l] < after);
        bool sm = valley || (l == 0);
        bool em = valley || (l == Ll - 1);
        if (sm) { ++total; if (cs < MAXC) starts[b * MAXC + cs++] = l; }
        if (em) { if (ce < MAXC) { int e = l + 2; ends[b * MAXC + ce++] = e > Ll ? Ll : e; } }
    }
    for (int m = cs; m < MAXC; ++m) starts[b * MAXC + m] = Ll;   // empty segment
    for (int m = ce; m < MAXC; ++m) ends[b * MAXC + m]   = Ll;
    countsOut[b] = (float)total;
}

// =====================================================================
// Kernel 4b: out[b,m,:] = (sum_{l in seg} u[l]*x[l,:]) / max(sum u[l], 1e-6)
// grid = (MAXC, B), 256 threads (one per feature)
// =====================================================================
__global__ void uma_segavg(const float* __restrict__ xs,
                           const float* __restrict__ U,
                           const int* __restrict__ starts, const int* __restrict__ ends,
                           float* __restrict__ out) {
    const int m = blockIdx.x, b = blockIdx.y, d = threadIdx.x;
    const int s = starts[b * MAXC + m];
    const int e = ends[b * MAXC + m];
    float num = 0.f, den = 0.f;
    for (int l = s; l < e; ++l) {
        float u = U[(size_t)b * Ll + l];
        den += u;
        num += u * xs[((size_t)b * Ll + l) * Dd + d];
    }
    out[((size_t)b * MAXC + m) * Dd + d] = num / fmaxf(den, 1e-6f);
}

// =====================================================================
extern "C" void kernel_launch(void* const* d_in, const int* in_sizes, int n_in,
                              void* d_out, int out_size, void* d_ws, size_t ws_size,
                              hipStream_t stream) {
    const float* xs  = (const float*)d_in[0];
    const int*   ol  = (const int*)  d_in[1];
    const float* wq  = (const float*)d_in[2];
    const float* bq  = (const float*)d_in[3];
    const float* wk  = (const float*)d_in[4];
    const float* bk  = (const float*)d_in[5];
    const float* gw  = (const float*)d_in[8];
    float* out = (float*)d_out;
    float* countsOut = out + (size_t)Bb * MAXC * Dd;

    char* ws = (char*)d_ws;
    size_t off = 0;
    bf16_t* Xb  = (bf16_t*)(ws + off); off += (size_t)Bb * Ll * Dd * 2;
    bf16_t* Qb  = (bf16_t*)(ws + off); off += (size_t)Bb * Ll * Dd * 2;
    bf16_t* Kb  = (bf16_t*)(ws + off); off += (size_t)Bb * Ll * Dd * 2;
    bf16_t* Wqb = (bf16_t*)(ws + off); off += (size_t)Dd * Dd * 2;
    bf16_t* Wkb = (bf16_t*)(ws + off); off += (size_t)Dd * Dd * 2;
    float* S  = (float*)(ws + off); off += (size_t)Bb * Ll * 4;
    float* U  = (float*)(ws + off); off += (size_t)Bb * Ll * 4;
    int* starts = (int*)(ws + off); off += (size_t)Bb * MAXC * 4;
    int* ends   = (int*)(ws + off); off += (size_t)Bb * MAXC * 4;

    // one-time f32 -> bf16 conversions (4 elements / thread)
    {
        int n4 = (Bb * Ll * Dd) / 4;
        cvt_to_bf16<<<(n4 + 255) / 256, 256, 0, stream>>>(xs, Xb, n4);
        int w4 = (Dd * Dd) / 4;
        cvt_to_bf16<<<(w4 + 255) / 256, 256, 0, stream>>>(wq, Wqb, w4);
        cvt_to_bf16<<<(w4 + 255) / 256, 256, 0, stream>>>(wk, Wkb, w4);
    }

    uma_proj_qk<<<(Bb * Ll) / 16, 256, 0, stream>>>(Xb, Wqb, bq, Wkb, bk, Qb, Kb);
    uma_rowsum <<<Bb * (Ll / 16), 256, 0, stream>>>(Qb, Kb, ol, gw, S);
    uma_colsum <<<Bb * (Ll / 16), 256, 0, stream>>>(Qb, Kb, ol, gw, S, U);
    uma_valleys<<<Bb, 32, 0, stream>>>(U, starts, ends, countsOut);
    uma_segavg <<<dim3(MAXC, Bb), Dd, 0, stream>>>(xs, U, starts, ends, out);
}
